// BettiGCN_14456859918546
// MI455X (gfx1250) — matrix operand order repaired and checked by reading, not compiled, not run on hardware
//
#include <hip/hip_runtime.h>
#include <hip/hip_bf16.h>

typedef __attribute__((ext_vector_type(16))) _Float16 v16h;
typedef __attribute__((ext_vector_type(8)))  _Float16 v8h;
typedef __attribute__((ext_vector_type(8)))  float    v8f;

// ---------------------------------------------------------------------------
// Zero a float buffer
// ---------------------------------------------------------------------------
__global__ void zero_f32_kernel(float* __restrict__ p, long n) {
    long t = (long)blockIdx.x * blockDim.x + threadIdx.x;
    if (t < n) p[t] = 0.0f;
}

// ---------------------------------------------------------------------------
// Pack x (f32 [N,10]) -> xh (f16 [N,32], zero-padded K)
// ---------------------------------------------------------------------------
__global__ void pack_x_kernel(const float* __restrict__ x, _Float16* __restrict__ xh,
                              int n, int inCh) {
    long t = (long)blockIdx.x * blockDim.x + threadIdx.x;
    long total = (long)n * 32;
    if (t >= total) return;
    int col = (int)(t & 31);
    int i   = (int)(t >> 5);
    xh[t] = (col < inCh) ? (_Float16)x[(long)i * inCh + col] : (_Float16)0.f;
}

// ---------------------------------------------------------------------------
// Pack weights W (f32 [K,N]) -> Bp (f16, WMMA-ready):
//   Bp[kb*32*N + col*32 + j] = W[kb*32+j][col]   (zero-padded beyond K)
// Each lane's 16 B-fragment values become 32 contiguous bytes.
// ---------------------------------------------------------------------------
__global__ void pack_w_kernel(const float* __restrict__ W, _Float16* __restrict__ Bp,
                              int K, int N, int nkb) {
    long t = (long)blockIdx.x * blockDim.x + threadIdx.x;
    long total = (long)nkb * N * 32;
    if (t >= total) return;
    int j   = (int)(t & 31);
    int col = (int)((t >> 5) % N);
    int kb  = (int)(t / ((long)N * 32));
    int k   = kb * 32 + j;
    Bp[t] = (k < K) ? (_Float16)W[(long)k * N + col] : (_Float16)0.f;
}

// ---------------------------------------------------------------------------
// Degree
// ---------------------------------------------------------------------------
__global__ void edge_deg_kernel(const long long* __restrict__ ei, int E,
                                float* __restrict__ deg) {
    int e = blockIdx.x * blockDim.x + threadIdx.x;
    if (e < E) atomicAdd(&deg[(int)ei[(long)E + e]], 1.0f);
}

__global__ void deg_finalize_kernel(const float* __restrict__ deg,
                                    float* __restrict__ dinv,
                                    float* __restrict__ invdeg, int n) {
    int i = blockIdx.x * blockDim.x + threadIdx.x;
    if (i < n) {
        float d = deg[i] + 1.0f;           // +1 self loop
        dinv[i]   = rsqrtf(d);
        invdeg[i] = 1.0f / d;
    }
}

// ---------------------------------------------------------------------------
// WMMA GEMM:  C[M,N](f32) = Ah[M, nkb*32](f16 row-major) x Bp(f16 packed)
// One wave per 16x16 tile; branch-free K loop (K pre-padded to mult of 32).
// Per k-block: a = two contiguous 16B loads, b = one contiguous 32B load.
// k order: contiguous, k = 16*hi + s, same in A and B.
// ---------------------------------------------------------------------------
__global__ void gcn_wmma_gemm_kernel(const _Float16* __restrict__ Ah,
                                     const _Float16* __restrict__ Bp,
                                     float* __restrict__ C,
                                     int M, int N, int nkb) {
    const int lane = threadIdx.x & 31;
    const int wave = threadIdx.x >> 5;
    const int wavesPerBlock = blockDim.x >> 5;
    const int ntn = N >> 4;
    const int total_tiles = (M >> 4) * ntn;
    int tile = blockIdx.x * wavesPerBlock + wave;
    if (tile >= total_tiles) return;          // wave-uniform

    const int tm  = tile / ntn;
    const int tn  = tile - tm * ntn;
    const int l16 = lane & 15;
    const int hi  = lane >> 4;                 // 0 or 1
    const int row = tm * 16 + l16;
    const int col = tn * 16 + l16;
    const long lda = (long)nkb * 32;

    const _Float16* __restrict__ pa = Ah + (long)row * lda + hi * 16;
    const _Float16* __restrict__ pb = Bp + (long)col * 32 + hi * 16;
    const long bStep = (long)N * 32;

    v8f acc = {};
#pragma unroll 2
    for (int kb = 0; kb < nkb; ++kb) {
        v8h a0 = *(const v8h*)(pa);            // K {ka..ka+7}
        v8h a1 = *(const v8h*)(pa + 8);        // K {ka+8..ka+15}
        v16h b = *(const v16h*)(pb);           // 16 contiguous packed values
        v16h a;
#pragma unroll
        for (int i = 0; i < 8; ++i) { a[i] = a0[i]; a[8 + i] = a1[i]; }
        acc = __builtin_amdgcn_wmma_f32_16x16x32_f16(false, a, false, b,
                                                     (short)0, acc, false, false);
        pa += 32;
        pb += bStep;
    }

    const int rbase = tm * 16 + hi * 8;
    float* __restrict__ pc = C + (long)rbase * N + col;
#pragma unroll
    for (int r = 0; r < 8; ++r) { *pc = acc[r]; pc += N; }
}

// ---------------------------------------------------------------------------
// Edge scatter: agg[dst] += h[src] * dinv[src]*dinv[dst]   (float4 per thread)
// ---------------------------------------------------------------------------
__global__ void edge_scatter_kernel(const float* __restrict__ h,
                                    float* __restrict__ agg,
                                    const long long* __restrict__ ei,
                                    const float* __restrict__ dinv,
                                    int E, int cShift) {
    const int c4Shift = cShift - 2;
    long t = (long)blockIdx.x * blockDim.x + threadIdx.x;
    long total = ((long)E) << c4Shift;
    if (t >= total) return;
    int e = (int)(t >> c4Shift);
    int c = (int)(t & ((1 << c4Shift) - 1)) << 2;
    int s = (int)ei[e];
    int d = (int)ei[(long)E + e];
    float nrm = dinv[s] * dinv[d];
    const float4 v = *(const float4*)(h + (((long)s) << cShift) + c);
    float* dst = agg + (((long)d) << cShift) + c;
    atomicAdd(dst + 0, v.x * nrm);
    atomicAdd(dst + 1, v.y * nrm);
    atomicAdd(dst + 2, v.z * nrm);
    atomicAdd(dst + 3, v.w * nrm);
}

// ---------------------------------------------------------------------------
// Combine: v = agg + h*invdeg[node] + bias[c]; optional relu.
// Writes f32 (in place into agg) and, if ah != nullptr, an f16 copy that
// feeds the next WMMA GEMM as its A operand.
// ---------------------------------------------------------------------------
__global__ void combine_kernel(float* __restrict__ agg,
                               const float* __restrict__ h,
                               const float* __restrict__ invdeg,
                               const float* __restrict__ bias,
                               _Float16* __restrict__ ah,
                               int n, int cShift, int do_relu) {
    long t = (long)blockIdx.x * blockDim.x + threadIdx.x;
    long total = ((long)n) << cShift;
    if (t >= total) return;
    int i = (int)(t >> cShift);
    int c = (int)(t & ((1 << cShift) - 1));
    float v = agg[t] + h[t] * invdeg[i] + bias[c];
    if (do_relu) v = fmaxf(v, 0.0f);
    agg[t] = v;
    if (ah) ah[t] = (_Float16)v;
}

// ---------------------------------------------------------------------------
// Per-graph sum pool (+ node counts)
// ---------------------------------------------------------------------------
__global__ void pool_sum_kernel(const float* __restrict__ a,
                                const long long* __restrict__ batch,
                                float* __restrict__ pooled,
                                float* __restrict__ cnts,
                                int n, int cShift) {
    long t = (long)blockIdx.x * blockDim.x + threadIdx.x;
    long total = ((long)n) << cShift;
    if (t >= total) return;
    int i = (int)(t >> cShift);
    int c = (int)(t & ((1 << cShift) - 1));
    int g = (int)batch[i];
    atomicAdd(&pooled[(g << cShift) + c], a[t]);
    if (c == 0) atomicAdd(&cnts[g], 1.0f);
}

// ---------------------------------------------------------------------------
// Final MLP: out[g] = relu(mean_pool @ fcW1 + fcb1) @ fcW2 + fcb2
// ---------------------------------------------------------------------------
__global__ void final_fc_kernel(const float* __restrict__ pooled,
                                const float* __restrict__ cnts,
                                const float* __restrict__ fcW1,
                                const float* __restrict__ fcb1,
                                const float* __restrict__ fcW2,
                                const float* __restrict__ fcb2,
                                float* __restrict__ out, int C, int H) {
    const int g = blockIdx.x;
    const int j = threadIdx.x;                 // 0..H-1
    __shared__ float sh[64];
    float invcnt = 1.0f / fmaxf(cnts[g], 1.0f);
    float acc = fcb1[j];
    for (int k = 0; k < C; ++k)
        acc += (pooled[g * C + k] * invcnt) * fcW1[k * H + j];
    float hj = fmaxf(acc, 0.0f);
    sh[j] = hj * fcW2[j];
    __syncthreads();
    for (int s = 32; s > 0; s >>= 1) {
        if (j < s) sh[j] += sh[j + s];
        __syncthreads();
    }
    if (j == 0) out[g] = sh[0] + fcb2[0];
}

// ---------------------------------------------------------------------------
// Host orchestration
// ---------------------------------------------------------------------------
extern "C" void kernel_launch(void* const* d_in, const int* in_sizes, int n_in,
                              void* d_out, int out_size, void* d_ws, size_t ws_size,
                              hipStream_t stream) {
    const float*     x     = (const float*)d_in[0];
    const long long* ei    = (const long long*)d_in[1];   // int64 [2,E]
    const long long* batch = (const long long*)d_in[2];   // int64 [N]
    const float* W1   = (const float*)d_in[3];
    const float* b1   = (const float*)d_in[4];
    const float* W2   = (const float*)d_in[5];
    const float* b2   = (const float*)d_in[6];
    const float* W3   = (const float*)d_in[7];
    const float* b3   = (const float*)d_in[8];
    const float* fcW1 = (const float*)d_in[9];
    const float* fcb1 = (const float*)d_in[10];
    const float* fcW2 = (const float*)d_in[11];
    const float* fcb2 = (const float*)d_in[12];
    float* out = (float*)d_out;

    const int IN_CH = 10, HID = 64, C2 = 128;
    const int n = in_sizes[0] / IN_CH;        // 100000
    const int E = in_sizes[1] / 2;            // 3200000
    const int G = out_size;                   // 64

    // workspace carve-out (256B aligned)
    char* ws = (char*)d_ws;
    size_t off = 0;
    auto carve = [&](size_t bytes) -> void* {
        void* p = (void*)(ws + off);
        off = (off + bytes + 255) & ~(size_t)255;
        return p;
    };
    float*     deg    = (float*)carve((size_t)n * 4);
    float*     dinv   = (float*)carve((size_t)n * 4);
    float*     invdeg = (float*)carve((size_t)n * 4);
    float*     h      = (float*)carve((size_t)n * C2 * 4);     // GEMM output
    float*     agg    = (float*)carve((size_t)n * C2 * 4);     // aggregation (f32)
    _Float16*  ah     = (_Float16*)carve((size_t)n * C2 * 2);  // f16 activations
    _Float16*  xh     = (_Float16*)carve((size_t)n * 32 * 2);  // packed input (K->32)
    _Float16*  Bp1    = (_Float16*)carve((size_t)32  * HID * 2);
    _Float16*  Bp2    = (_Float16*)carve((size_t)HID * C2  * 2);
    _Float16*  Bp3    = (_Float16*)carve((size_t)C2  * C2  * 2);
    float*     pooled = (float*)carve((size_t)G * C2 * 4);
    float*     cnts   = (float*)carve((size_t)G * 4);
    (void)ws_size; (void)n_in;

    const int T = 256;
    auto blocks = [&](long total) { return (int)((total + T - 1) / T); };

    // ---- one-time packs (cheap, every call for determinism) ----
    pack_x_kernel<<<blocks((long)n * 32), T, 0, stream>>>(x, xh, n, IN_CH);
    pack_w_kernel<<<blocks((long)1 * HID * 32), T, 0, stream>>>(W1, Bp1, IN_CH, HID, 1);
    pack_w_kernel<<<blocks((long)2 * C2 * 32),  T, 0, stream>>>(W2, Bp2, HID,  C2, 2);
    pack_w_kernel<<<blocks((long)4 * C2 * 32),  T, 0, stream>>>(W3, Bp3, C2,   C2, 4);

    // ---- degree (shared by all 3 layers) ----
    zero_f32_kernel<<<blocks(n), T, 0, stream>>>(deg, n);
    edge_deg_kernel<<<blocks(E), T, 0, stream>>>(ei, E, deg);
    deg_finalize_kernel<<<blocks(n), T, 0, stream>>>(deg, dinv, invdeg, n);

    auto gemm = [&](const _Float16* A, const _Float16* Bp, float* Cp,
                    int M, int N, int nkb) {
        int tiles = (M / 16) * (N / 16);
        int wavesPerBlock = 4;                       // 128 threads
        int gb = (tiles + wavesPerBlock - 1) / wavesPerBlock;
        gcn_wmma_gemm_kernel<<<gb, wavesPerBlock * 32, 0, stream>>>(A, Bp, Cp, M, N, nkb);
    };

    auto layer = [&](const _Float16* inA, int nkb, const _Float16* Bp,
                     const float* bias, int Cdim, int cShift,
                     int do_relu, _Float16* ahOut) {
        gemm(inA, Bp, h, n, Cdim, nkb);                                  // h = A @ W
        zero_f32_kernel<<<blocks((long)n * Cdim), T, 0, stream>>>(agg, (long)n * Cdim);
        edge_scatter_kernel<<<blocks(((long)E) * (Cdim / 4)), T, 0, stream>>>(
            h, agg, ei, dinv, E, cShift);
        combine_kernel<<<blocks((long)n * Cdim), T, 0, stream>>>(
            agg, h, invdeg, bias, ahOut, n, cShift, do_relu);            // f32 in agg
    };

    // ---- 3 GCN layers ----
    layer(xh, 1, Bp1, b1, HID, 6, 1, ah);        // agg/ah <- relu(gcn1)   [n,64]
    layer(ah, 2, Bp2, b2, C2,  7, 1, ah);        // agg/ah <- relu(gcn2)   [n,128]
    layer(ah, 4, Bp3, b3, C2,  7, 0, nullptr);   // agg    <- gcn3         [n,128]

    // ---- global mean pool + MLP head ----
    zero_f32_kernel<<<blocks((long)G * C2), T, 0, stream>>>(pooled, (long)G * C2);
    zero_f32_kernel<<<1, 64, 0, stream>>>(cnts, G);
    pool_sum_kernel<<<blocks((long)n * C2), T, 0, stream>>>(agg, batch, pooled, cnts, n, 7);
    final_fc_kernel<<<G, HID, 0, stream>>>(pooled, cnts, fcW1, fcb1, fcW2, fcb2,
                                           out, C2, HID);
}
